// AttentionBlock_26817775796754
// MI455X (gfx1250) — matrix operand order repaired and checked by reading, not compiled, not run on hardware
//
#include <hip/hip_runtime.h>
#include <hip/hip_bf16.h>

typedef _Float16 f16;
typedef __attribute__((ext_vector_type(16))) _Float16 v16h;
typedef __attribute__((ext_vector_type(8)))  _Float16 v8h;
typedef __attribute__((ext_vector_type(8)))  float    v8f;

#define B_   4
#define C_   512
#define N_   2048
#define H_   8
#define DK_  64
#define QKVC 1536          // 3*H*DK
#define MTOT (B_ * N_)     // 8192

// ---------- WMMA fragment helpers (contiguous-K order used by this file) ----------

// A-matrix 16x32 f16: lane m = lane&15 supplies row M=m.
// lanes 0-15: K = 0..15; lanes 16-31: K = 16..31 (contiguous K).
static __device__ __forceinline__ v16h load_a16(const f16* row, bool hi) {
  return *(const v16h*)(row + (hi ? 16 : 0));
}

// B-matrix 32x16 f16: lane supplies column N=lane&15;
// lanes 0-15 hold K=0..15, lanes 16-31 hold K=16..31 (contiguous K).
static __device__ __forceinline__ v16h load_b16(const f16* rowN, bool hi) {
  return *(const v16h*)(rowN + (hi ? 16 : 0));
}

static __device__ __forceinline__ v8f wmma_f16(v16h a, v16h b, v8f c) {
  return __builtin_amdgcn_wmma_f32_16x16x32_f16(false, a, false, b,
                                                (short)0, c, false, false);
}

// Async global->LDS 16B copy (CDNA5 GLOBAL_LOAD_ASYNC_TO_LDS_B128, ASYNCcnt).
static __device__ __forceinline__ void async_ld_b128(unsigned lds_addr,
                                                     unsigned goff,
                                                     const void* sbase) {
  asm volatile("global_load_async_to_lds_b128 %0, %1, %2"
               :
               : "v"(lds_addr), "v"(goff), "s"(sbase)
               : "memory");
}
static __device__ __forceinline__ void wait_async0() {
  asm volatile("s_wait_asynccnt 0x0" ::: "memory");
}

// ---------- shared GEMM micro-step (32x64 tile per wave) -------------------
static __device__ __forceinline__ void gemm_fetch(
    v16h A[2], v16h Bq[4],
    const f16* ar0, const f16* ar1,
    const f16* br0, const f16* br1, const f16* br2, const f16* br3,
    int k, bool hi) {
  A[0]  = load_a16(ar0 + k, hi);
  A[1]  = load_a16(ar1 + k, hi);
  Bq[0] = load_b16(br0 + k, hi);
  Bq[1] = load_b16(br1 + k, hi);
  Bq[2] = load_b16(br2 + k, hi);
  Bq[3] = load_b16(br3 + k, hi);
}

static __device__ __forceinline__ void gemm_step(v8f acc[2][4],
                                                 const v16h A[2],
                                                 const v16h Bq[4]) {
#pragma unroll
  for (int r = 0; r < 2; ++r)
#pragma unroll
    for (int t = 0; t < 4; ++t)
      acc[r][t] = wmma_f16(A[r], Bq[t], acc[r][t]);
}

// ---------- Kernel 0a: transpose x [B,C,N] f32 -> xt [B*N, C] f16 ----------
__global__ __launch_bounds__(256) void xpose_f32_to_f16(
    const float* __restrict__ x, f16* __restrict__ xt) {
  __shared__ float tile[32][33];
  const int b  = blockIdx.z;
  const int n0 = blockIdx.x * 32;
  const int c0 = blockIdx.y * 32;
  const int tx = threadIdx.x, ty = threadIdx.y;   // 32 x 8
#pragma unroll
  for (int k = 0; k < 4; ++k) {
    int c = c0 + ty + k * 8;
    tile[ty + k * 8][tx] = x[((size_t)b * C_ + c) * N_ + n0 + tx];
  }
  __syncthreads();
#pragma unroll
  for (int k = 0; k < 4; ++k) {
    int n = n0 + ty + k * 8;
    xt[((size_t)b * N_ + n) * C_ + c0 + tx] = (f16)tile[tx][ty + k * 8];
  }
}

// ---------- Kernel 0b: f32 -> f16 elementwise (weights) ----------
__global__ __launch_bounds__(256) void cvt_f32_f16(
    const float* __restrict__ s, f16* __restrict__ d, int n) {
  int i = blockIdx.x * blockDim.x + threadIdx.x;
  const int stride = gridDim.x * blockDim.x;
  for (; i < n; i += stride) d[i] = (f16)s[i];
}

// ---------- Kernel 1: QKV projection GEMM (ping-pong pipelined) ------------
__global__ __launch_bounds__(128) void qkv_gemm(
    const f16* __restrict__ xt, const f16* __restrict__ Wp,
    const float* __restrict__ bp,
    f16* __restrict__ Q, f16* __restrict__ K, f16* __restrict__ V) {
  const int lane = threadIdx.x & 31;
  const int wave = threadIdx.x >> 5;
  const int m    = lane & 15;
  const bool hi  = lane >= 16;
  const int n0   = blockIdx.x * 64;               // output channel group
  const int i0   = blockIdx.y * 128 + wave * 32;  // row group over b*n
  v8f acc[2][4] = {};
  const f16* ar0 = xt + (size_t)(i0 + m) * C_;
  const f16* ar1 = xt + (size_t)(i0 + 16 + m) * C_;
  const f16* br0 = Wp + (size_t)(n0 +  0 + m) * C_;
  const f16* br1 = Wp + (size_t)(n0 + 16 + m) * C_;
  const f16* br2 = Wp + (size_t)(n0 + 32 + m) * C_;
  const f16* br3 = Wp + (size_t)(n0 + 48 + m) * C_;
  v16h Af[2][2], Bf[2][4];
  gemm_fetch(Af[0], Bf[0], ar0, ar1, br0, br1, br2, br3, 0, hi);
#pragma unroll 1
  for (int kb = 0; kb < C_; kb += 64) {
    gemm_fetch(Af[1], Bf[1], ar0, ar1, br0, br1, br2, br3, kb + 32, hi);
    gemm_step(acc, Af[0], Bf[0]);
    const int pk = (kb + 64 < C_) ? kb + 64 : 0;   // branchless wrap prefetch
    gemm_fetch(Af[0], Bf[0], ar0, ar1, br0, br1, br2, br3, pk, hi);
    gemm_step(acc, Af[1], Bf[1]);
  }
#pragma unroll
  for (int rt = 0; rt < 2; ++rt) {
#pragma unroll
    for (int t = 0; t < 4; ++t) {
      const int cc   = n0 + t * 16 + m;
      const int h    = cc / 192;
      const int r    = cc % 192;
      const int kind = r >> 6;          // 0=q 1=k 2=v
      const int d    = r & 63;
      const float bias = bp[cc];
      f16* base = (kind == 0) ? Q : (kind == 1) ? K : V;
      const float sc = (kind == 0) ? 0.125f : 1.0f;   // DK^-0.5 folded into Q
#pragma unroll
      for (int v = 0; v < 8; ++v) {
        const int g  = i0 + rt * 16 + v + (hi ? 8 : 0);
        const int b_ = g >> 11;
        const int nn = g & (N_ - 1);
        base[(((size_t)b_ * H_ + h) * N_ + nn) * DK_ + d] =
            (f16)((acc[rt][t][v] + bias) * sc);
      }
    }
  }
}

// ---------- Kernel 2: per-column softmax stats (softmax over query axis) ----
__global__ __launch_bounds__(128) void col_softmax_stats(
    const f16* __restrict__ Q, const f16* __restrict__ K,
    float* __restrict__ mArr, float* __restrict__ zArr) {
  const int lane = threadIdx.x & 31;
  const int wave = threadIdx.x >> 5;
  const int m    = lane & 15;
  const bool hi  = lane >= 16;
  const int bh   = blockIdx.x;
  const int j0   = blockIdx.y * 64 + wave * 16;
  const f16* Qb = Q + (size_t)bh * N_ * DK_;
  const f16* Kb = K + (size_t)bh * N_ * DK_;
  // B frags fixed for the whole i loop (N=j columns, K=d contiguous)
  const v16h bk0 = load_b16(Kb + (size_t)(j0 + m) * DK_ + 0,  hi);
  const v16h bk1 = load_b16(Kb + (size_t)(j0 + m) * DK_ + 32, hi);
  float rm = -3.0e38f, rz = 0.f;
  v16h A[2][2];
  A[0][0] = load_a16(Qb + (size_t)m * DK_ + 0,  hi);
  A[0][1] = load_a16(Qb + (size_t)m * DK_ + 32, hi);
#pragma unroll 1
  for (int i0 = 0; i0 < N_; i0 += 32) {
#pragma unroll
    for (int half = 0; half < 2; ++half) {
      const int cur = half;
      const int nxt = half ^ 1;
      const int pi0 = (half == 0) ? (i0 + 16)
                                  : ((i0 + 32 < N_) ? i0 + 32 : 0);
      A[nxt][0] = load_a16(Qb + (size_t)(pi0 + m) * DK_ + 0,  hi);
      A[nxt][1] = load_a16(Qb + (size_t)(pi0 + m) * DK_ + 32, hi);
      v8f s = {};
      s = wmma_f16(A[cur][0], bk0, s);
      s = wmma_f16(A[cur][1], bk1, s);
      float tmax = s[0];
#pragma unroll
      for (int v = 1; v < 8; ++v) tmax = fmaxf(tmax, s[v]);
      const float nm = fmaxf(rm, tmax);
      float acc = 0.f;
#pragma unroll
      for (int v = 0; v < 8; ++v) acc += __expf(s[v] - nm);
      rz = rz * __expf(rm - nm) + acc;
      rm = nm;
    }
  }
  // lanes L and L+16 hold the same column j: combine
  const float om = __shfl_xor(rm, 16, 32);
  const float oz = __shfl_xor(rz, 16, 32);
  const float fm = fmaxf(rm, om);
  const float fz = rz * __expf(rm - fm) + oz * __expf(om - fm);
  if (lane < 16) {
    mArr[(size_t)bh * N_ + j0 + lane] = fm;
    zArr[(size_t)bh * N_ + j0 + lane] = 1.0f / fz;
  }
}

// ---------- Kernel 3: O = exp(S - m[j]) * (V[j]/Z[j]), streamed over j ------
// K/V tiles double-buffered in LDS via GLOBAL_LOAD_ASYNC_TO_LDS_B128:
// tile i+1 is fetched asynchronously while tile i feeds the WMMAs.
__global__ __launch_bounds__(128) void attn_av(
    const f16* __restrict__ Q, const f16* __restrict__ K,
    const f16* __restrict__ V,
    const float* __restrict__ mArr, const float* __restrict__ zArr,
    f16* __restrict__ res) {
  __shared__ __align__(32) f16 tK[2][32][64];   // K tiles, K-dim contiguous
  __shared__ __align__(32) f16 tV[2][32][72];   // V tiles, padded rows
  __shared__ __align__(32) f16 tP[4][16][32];   // per-wave P re-layout buffer
  __shared__ float sM[N_];
  __shared__ float sZ[N_];
  const int tid  = threadIdx.x;
  const int lane = tid & 31;
  const int wave = tid >> 5;
  const int m    = lane & 15;
  const bool hi  = lane >= 16;
  const int bh = blockIdx.x;
  const int b_ = bh / H_;
  const int h  = bh % H_;
  const int i0 = blockIdx.y * 64 + wave * 16;
  const f16* Qb = Q + (size_t)bh * N_ * DK_;
  const f16* Kb = K + (size_t)bh * N_ * DK_;
  const f16* Vb = V + (size_t)bh * N_ * DK_;

  auto stage = [&](int jb, int buf) {
#pragma unroll
    for (int q = 0; q < 2; ++q) {           // K tile: 256 x 16B chunks
      const int u  = tid + q * 128;
      const int r  = u >> 3;
      const int ch = (u & 7) * 8;
      async_ld_b128((unsigned)(uintptr_t)&tK[buf][r][ch],
                    (unsigned)(((jb + r) * DK_ + ch) * sizeof(f16)), Kb);
    }
#pragma unroll
    for (int q = 0; q < 2; ++q) {           // V tile (padded LDS rows)
      const int u  = tid + q * 128;
      const int r  = u >> 3;
      const int ch = (u & 7) * 8;
      async_ld_b128((unsigned)(uintptr_t)&tV[buf][r][ch],
                    (unsigned)(((jb + r) * DK_ + ch) * sizeof(f16)), Vb);
    }
  };

  for (int idx = tid; idx < N_; idx += 128) {
    sM[idx] = mArr[(size_t)bh * N_ + idx];
    sZ[idx] = zArr[(size_t)bh * N_ + idx];
  }
  stage(0, 0);                              // prologue: tile 0 in flight
  // Q fragments for this wave's 16 rows: registers for the whole j loop
  const v16h aq0 = load_a16(Qb + (size_t)(i0 + m) * DK_ + 0,  hi);
  const v16h aq1 = load_a16(Qb + (size_t)(i0 + m) * DK_ + 32, hi);
  v8f o[4] = {};
  wait_async0();
  __syncthreads();                          // tile 0 + sM/sZ visible

#pragma unroll 1
  for (int jb = 0; jb < N_; jb += 32) {
    const int cb  = (jb >> 5) & 1;
    const int pjb = (jb + 32 < N_) ? jb + 32 : 0;   // branchless wrap prefetch
    stage(pjb, cb ^ 1);                     // async fetch of next tile
    // ---- S tiles for two 16-wide j subtiles; P -> LDS in A-layout order ----
#pragma unroll
    for (int st = 0; st < 2; ++st) {
      v8f s = {};
      v16h bk0 = load_b16(&tK[cb][st * 16 + m][0],  hi);
      v16h bk1 = load_b16(&tK[cb][st * 16 + m][32], hi);
      s = wmma_f16(aq0, bk0, s);
      s = wmma_f16(aq1, bk1, s);
      const int j = jb + st * 16 + m;       // this lane's column
      const float mj = sM[j], rz = sZ[j];
#pragma unroll
      for (int v = 0; v < 8; ++v) {
        const float p = __expf(s[v] - mj) * rz;
        tP[wave][v + (hi ? 8 : 0)][st * 16 + m] = (f16)p;
      }
    }
    // ---- P as A fragment (16i x 32j), V tile as B (32j x 16d), 4 d subtiles
    const v16h ap = load_a16(&tP[wave][m][0], hi);
#pragma unroll
    for (int t = 0; t < 4; ++t) {
      const f16* vp = &tV[cb][hi ? 16 : 0][t * 16 + m];
      v16h bv;
#pragma unroll
      for (int e = 0; e < 16; ++e) bv[e] = vp[(size_t)e * 72];
      o[t] = wmma_f16(ap, bv, o[t]);
    }
    wait_async0();                          // next tile landed
    __syncthreads();                        // all waves done with current tile
  }
  // write res [b*n, H*DK] f16 (head-interleaved channels for the out GEMM)
#pragma unroll
  for (int t = 0; t < 4; ++t) {
    const int d = t * 16 + m;
#pragma unroll
    for (int v = 0; v < 8; ++v) {
      const int i = i0 + v + (hi ? 8 : 0);
      res[((size_t)b_ * N_ + i) * C_ + h * DK_ + d] = (f16)o[t][v];
    }
  }
}

// ---------- Kernel 4: out = res @ Wo^T + bo + x, stored transposed [B,C,N] -
__global__ __launch_bounds__(128) void out_gemm(
    const f16* __restrict__ res, const f16* __restrict__ Wo,
    const float* __restrict__ bo, const float* __restrict__ x,
    float* __restrict__ out) {
  const int lane = threadIdx.x & 31;
  const int wave = threadIdx.x >> 5;
  const int m    = lane & 15;
  const bool hi  = lane >= 16;
  const int c0 = blockIdx.x * 64;
  const int i0 = blockIdx.y * 128 + wave * 32;
  v8f acc[2][4] = {};
  const f16* ar0 = res + (size_t)(i0 + m) * C_;
  const f16* ar1 = res + (size_t)(i0 + 16 + m) * C_;
  const f16* br0 = Wo + (size_t)(c0 +  0 + m) * C_;
  const f16* br1 = Wo + (size_t)(c0 + 16 + m) * C_;
  const f16* br2 = Wo + (size_t)(c0 + 32 + m) * C_;
  const f16* br3 = Wo + (size_t)(c0 + 48 + m) * C_;
  v16h Af[2][2], Bf[2][4];
  gemm_fetch(Af[0], Bf[0], ar0, ar1, br0, br1, br2, br3, 0, hi);
#pragma unroll 1
  for (int kb = 0; kb < C_; kb += 64) {
    gemm_fetch(Af[1], Bf[1], ar0, ar1, br0, br1, br2, br3, kb + 32, hi);
    gemm_step(acc, Af[0], Bf[0]);
    const int pk = (kb + 64 < C_) ? kb + 64 : 0;
    gemm_fetch(Af[0], Bf[0], ar0, ar1, br0, br1, br2, br3, pk, hi);
    gemm_step(acc, Af[1], Bf[1]);
  }
#pragma unroll
  for (int rt = 0; rt < 2; ++rt) {
#pragma unroll
    for (int t = 0; t < 4; ++t) {
      const int c = c0 + t * 16 + m;
      const float bias = bo[c];
#pragma unroll
      for (int v = 0; v < 8; ++v) {
        const int g  = i0 + rt * 16 + v + (hi ? 8 : 0);
        const int b_ = g >> 11;
        const int nn = g & (N_ - 1);
        const size_t off = ((size_t)b_ * C_ + c) * N_ + nn;
        out[off] = acc[rt][t][v] + bias + x[off];   // fused bias + residual
      }
    }
  }
}

// ---------------------------- launcher ------------------------------------
extern "C" void kernel_launch(void* const* d_in, const int* in_sizes, int n_in,
                              void* d_out, int out_size, void* d_ws, size_t ws_size,
                              hipStream_t stream) {
  const float* x  = (const float*)d_in[0];  // [B,C,N]
  const float* Wp = (const float*)d_in[1];  // [1536,512]
  const float* bp = (const float*)d_in[2];  // [1536]
  const float* Wo = (const float*)d_in[3];  // [512,512]
  const float* bo = (const float*)d_in[4];  // [512]
  float* out = (float*)d_out;               // [B,C,N]

  char* ws = (char*)d_ws;
  size_t off = 0;
  auto carve = [&](size_t bytes) -> char* {
    char* p = ws + off;
    off = (off + bytes + 255) & ~(size_t)255;
    return p;
  };
  f16*   xt16  = (f16*)carve((size_t)MTOT * C_ * sizeof(f16));
  f16*   Wp16  = (f16*)carve((size_t)QKVC * C_ * sizeof(f16));
  f16*   Wo16  = (f16*)carve((size_t)C_ * C_ * sizeof(f16));
  f16*   Q16   = (f16*)carve((size_t)B_ * H_ * N_ * DK_ * sizeof(f16));
  f16*   K16   = (f16*)carve((size_t)B_ * H_ * N_ * DK_ * sizeof(f16));
  f16*   V16   = (f16*)carve((size_t)B_ * H_ * N_ * DK_ * sizeof(f16));
  float* mCol  = (float*)carve((size_t)B_ * H_ * N_ * sizeof(float));
  float* zCol  = (float*)carve((size_t)B_ * H_ * N_ * sizeof(float));
  f16*   res16 = (f16*)carve((size_t)MTOT * C_ * sizeof(f16));   // ~45 MB total

  {
    dim3 g(N_ / 32, C_ / 32, B_), b(32, 8);
    hipLaunchKernelGGL(xpose_f32_to_f16, g, b, 0, stream, x, xt16);
  }
  hipLaunchKernelGGL(cvt_f32_f16, dim3(512), dim3(256), 0, stream, Wp, Wp16, QKVC * C_);
  hipLaunchKernelGGL(cvt_f32_f16, dim3(256), dim3(256), 0, stream, Wo, Wo16, C_ * C_);
  hipLaunchKernelGGL(qkv_gemm, dim3(QKVC / 64, MTOT / 128), dim3(128), 0, stream,
                     xt16, Wp16, bp, Q16, K16, V16);
  hipLaunchKernelGGL(col_softmax_stats, dim3(B_ * H_, N_ / 64), dim3(128), 0, stream,
                     Q16, K16, mCol, zCol);
  hipLaunchKernelGGL(attn_av, dim3(B_ * H_, N_ / 64), dim3(128), 0, stream,
                     Q16, K16, V16, mCol, zCol, res16);
  hipLaunchKernelGGL(out_gemm, dim3(C_ / 64, MTOT / 128), dim3(128), 0, stream,
                     res16, Wo16, bo, x, out);
}
